// DualLossHopfieldLayer_72224170049546
// MI455X (gfx1250) — compile-verified, matrix operand order checked by reading, not run on hardware
//
#include <hip/hip_runtime.h>
#include <hip/hip_bf16.h>

typedef __attribute__((ext_vector_type(16))) __bf16 v16bf;
typedef __attribute__((ext_vector_type(8)))  float  v8f;
typedef __attribute__((ext_vector_type(4)))  float  f4;
typedef __attribute__((ext_vector_type(4)))  unsigned int u32x4;
typedef __attribute__((ext_vector_type(8)))  int  i32x8;
typedef __attribute__((ext_vector_type(4)))  int  i32x4;

#define NDOCS   100000
#define DMODEL  768
#define DPROJ   256
#define NHEADS  4
#define HEADDIM 64
#define BATCH   8
#define SEQ     2048
#define HID     2048

#define TILE_DOCS 32
#define A_STRIDE  (DMODEL + 8)
#define RAW_BYTES (TILE_DOCS * DMODEL * 4)
#define A_BYTES   (TILE_DOCS * A_STRIDE * 2)
#define KV_DYN_LDS (RAW_BYTES + A_BYTES)

#if __has_builtin(__builtin_amdgcn_tensor_load_to_lds)
#define HAVE_TDM 1
#else
#define HAVE_TDM 0
#endif

// ---------------------------------------------------------------------------
// Kernel 1: layernorm(query) -> xi0 = q @ Wq.T ; beta = exp(log_beta)
// ---------------------------------------------------------------------------
__global__ __launch_bounds__(256) void hop_prep(
    const float* __restrict__ qe, const float* __restrict__ Wq,
    const float* __restrict__ gq, const float* __restrict__ bq,
    const float* __restrict__ log_beta,
    float* __restrict__ xi0, float* __restrict__ beta_out)
{
  __shared__ float s_q[BATCH][DMODEL];
  __shared__ float s_ps[BATCH][32];
  __shared__ float s_pq[BATCH][32];
  __shared__ float s_mu[BATCH], s_rsd[BATCH];
  int tid = threadIdx.x;
  int row = tid >> 5, lane = tid & 31;
  const float* src = qe + row * DMODEL;
  float s = 0.f, ss = 0.f;
  for (int i = lane; i < DMODEL; i += 32) { float x = src[i]; s += x; ss += x * x; }
  s_ps[row][lane] = s; s_pq[row][lane] = ss;
  __syncthreads();
  if (lane == 0) {
    float S = 0.f, SS = 0.f;
    for (int i = 0; i < 32; i++) { S += s_ps[row][i]; SS += s_pq[row][i]; }
    float mu = S / DMODEL;
    float var = SS / DMODEL - mu * mu;
    s_mu[row] = mu; s_rsd[row] = rsqrtf(var + 1e-5f);
  }
  __syncthreads();
  float mu = s_mu[row], rs = s_rsd[row];
  for (int i = lane; i < DMODEL; i += 32) {
    float x = src[i];
    s_q[row][i] = (x - mu) * rs * gq[i] + bq[i];
  }
  __syncthreads();
  const float* wr = Wq + (size_t)tid * DMODEL;
  for (int b = 0; b < BATCH; b++) {
    float acc = 0.f;
    for (int c = 0; c < DMODEL; c++) acc += s_q[b][c] * wr[c];
    xi0[b * DPROJ + tid] = acc;
  }
  if (tid < NHEADS) beta_out[tid] = expf(log_beta[tid]);
}

// ---------------------------------------------------------------------------
// TDM helper: DMA one 32x768 f32 tile from global into LDS (per D# spec,
// CDNA5 ISA). tensor == tile (dims 768 x 32), so all
// accesses are in-bounds by construction.
// ---------------------------------------------------------------------------
#if HAVE_TDM
__device__ __forceinline__ void tdm_load_tile(const float* gsrc, unsigned lds_byte_off)
{
  unsigned long long ga = (unsigned long long)(const void*)gsrc;
  u32x4 g0;
  g0[0] = 1u;                                        // count=1, user mode
  g0[1] = lds_byte_off;                              // lds_addr (bytes)
  g0[2] = (unsigned)(ga & 0xFFFFFFFFu);              // global_addr[31:0]
  g0[3] = (unsigned)((ga >> 32) & 0x1FFFFFFu)        // global_addr[56:32]
        | (2u << 30);                                // type = 2 ("image")
  i32x8 g1;
  g1[0] = (int)(2u << 16);                           // data_size=4B, wg_mask=0
  g1[1] = (int)((unsigned)(DMODEL & 0xFFFF) << 16);  // tensor_dim0[15:0]
  g1[2] = (int)((unsigned)TILE_DOCS << 16);          // dim0[31:16]=0 | tensor_dim1[15:0]
  g1[3] = (int)((unsigned)DMODEL << 16);             // dim1[31:16]=0 | tile_dim0
  g1[4] = TILE_DOCS;                                 // tile_dim1 (tile_dim2=0)
  g1[5] = DMODEL;                                    // tensor_dim0_stride[31:0]
  g1[6] = 0;
  g1[7] = 0;
  i32x4 z4 = {0, 0, 0, 0};
#if defined(__clang_major__) && (__clang_major__ >= 23)
  i32x8 z8 = {0, 0, 0, 0, 0, 0, 0, 0};
  __builtin_amdgcn_tensor_load_to_lds(g0, g1, z4, z4, z8, 0);
#else
  __builtin_amdgcn_tensor_load_to_lds(g0, g1, z4, z4, 0);
#endif
}
#endif

// ---------------------------------------------------------------------------
// Kernel 2 (flagship): TDM-stage a 32x768 f32 tile into LDS, layernorm ->
// bf16 LDS, then K = mem@Wk.T and V = mem@Wv.T via v_wmma_f32_16x16x32_bf16.
// ---------------------------------------------------------------------------
__global__ __launch_bounds__(256) void hop_kv_project(
    const float* __restrict__ mem, const float* __restrict__ Wk,
    const float* __restrict__ Wv, const float* __restrict__ gm,
    const float* __restrict__ bm, void* Kb_, void* Vb_)
{
  extern __shared__ char dynsmem[];
  float*  raw = (float*)dynsmem;                     // 32 x 768 f32 (96 KB)
  __bf16* s_A = (__bf16*)(dynsmem + RAW_BYTES);      // 32 x 776 bf16 (48.5 KB)
  __shared__ float s_ps[TILE_DOCS][8], s_pq[TILE_DOCS][8];
  __shared__ float s_mu[TILE_DOCS], s_rsd[TILE_DOCS];
  __bf16* Kb = (__bf16*)Kb_;
  __bf16* Vb = (__bf16*)Vb_;

  int tid = threadIdx.x;
  int docBase = blockIdx.x * TILE_DOCS;

  // Phase 0: stage the raw tile into LDS (TDM DMA when available).
#if HAVE_TDM
  if (tid < 32) {
    unsigned dynBase;
#if __has_builtin(__builtin_amdgcn_groupstaticsize)
    dynBase = (unsigned)__builtin_amdgcn_groupstaticsize();
#else
    dynBase = 0u;
#endif
    tdm_load_tile(mem + (size_t)docBase * DMODEL, dynBase);
#if __has_builtin(__builtin_amdgcn_s_wait_tensorcnt)
    __builtin_amdgcn_s_wait_tensorcnt(0);
#else
    asm volatile("s_wait_tensorcnt 0x0" ::: "memory");
#endif
  }
  __syncthreads();
#else
  {
    const f4* gsrc = (const f4*)(mem + (size_t)docBase * DMODEL);
    f4* dst = (f4*)raw;
    for (int i = tid; i < TILE_DOCS * DMODEL / 4; i += 256) dst[i] = gsrc[i];
  }
  __syncthreads();
#endif

  // Phase 1: per-row layernorm (stats from LDS) -> bf16 into LDS
  {
    int row = tid >> 3, seg = tid & 7;
    const float* src = raw + row * DMODEL + seg * 96;
    float s = 0.f, ss = 0.f;
    for (int i = 0; i < 96; i++) { float x = src[i]; s += x; ss += x * x; }
    s_ps[row][seg] = s; s_pq[row][seg] = ss;
    __syncthreads();
    if (seg == 0) {
      float S = 0.f, SS = 0.f;
      for (int i = 0; i < 8; i++) { S += s_ps[row][i]; SS += s_pq[row][i]; }
      float mu = S / DMODEL;
      float var = SS / DMODEL - mu * mu;
      s_mu[row] = mu; s_rsd[row] = rsqrtf(var + 1e-5f);
    }
    __syncthreads();
    float mu = s_mu[row], rs = s_rsd[row];
    int cb = seg * 96;
    for (int i = 0; i < 96; i++) {
      int c = cb + i;
      float x = src[i];
      s_A[row * A_STRIDE + c] = (__bf16)((x - mu) * rs * gm[c] + bm[c]);
    }
  }
  __syncthreads();

  // Phase 2: WMMA. bf16 A 16x32 ISA layout: lanes 0-15 row M, K {0..7,16..23};
  // lanes 16-31 same M, K {8..15,24..31}. B is the N-symmetric layout.
  int wave = tid >> 5, lane = tid & 31;
  int l15 = lane & 15, hi = lane >> 4;
  for (int u = wave; u < 64; u += 8) {
    int isV = u >> 5, dt = (u >> 4) & 1, nt = u & 15;
    const float* W = isV ? Wv : Wk;
    __bf16* Out = isV ? Vb : Kb;
    int mrow = dt * 16 + l15;
    int ncol = nt * 16 + l15;
    const float* wrow = W + (size_t)ncol * DMODEL;   // B[k,n] = W[ncol, k]
    v8f acc = {0.f, 0.f, 0.f, 0.f, 0.f, 0.f, 0.f, 0.f};
    for (int kk = 0; kk < 24; kk++) {
      int k0 = kk * 32 + hi * 8;
      v16bf a, bmat;
      const __bf16* ap = &s_A[mrow * A_STRIDE + k0];
      const float*  bp = wrow + k0;
#pragma unroll
      for (int i = 0; i < 8; i++) {
        a[i]      = ap[i];
        a[i + 8]  = ap[i + 16];
        bmat[i]     = (__bf16)bp[i];
        bmat[i + 8] = (__bf16)bp[i + 16];
      }
      acc = __builtin_amdgcn_wmma_f32_16x16x32_bf16(
          false, a, false, bmat, (short)0, acc, false, false);
    }
    // C/D layout: lane<16 -> (M=j, N=lane); lane>=16 -> (M=8+j, N=lane-16)
    int mBase = dt * 16 + hi * 8;
#pragma unroll
    for (int j = 0; j < 8; j++) {
      Out[(size_t)(docBase + mBase + j) * DPROJ + ncol] = (__bf16)acc[j];
    }
  }
}

// ---------------------------------------------------------------------------
// Kernel 3: one Hopfield step for one (b,h) pair per workgroup.
// scores -> z (L2-resident ws), block max, 30-iter bisection for the
// entmax-1.5 threshold tau, deterministic sparse weighted V-sum.
// ---------------------------------------------------------------------------
__global__ __launch_bounds__(1024) void hop_step(
    const void* Kb_, const void* Vb_, const float* __restrict__ beta,
    const float* __restrict__ xi_in, float* __restrict__ xi_out,
    float* __restrict__ zbuf)
{
  const __bf16* Kb = (const __bf16*)Kb_;
  const __bf16* Vb = (const __bf16*)Vb_;
  __shared__ float s_xiq[HEADDIM];
  __shared__ float s_red[1024];
  __shared__ float s_part[16][HEADDIM];
  __shared__ float s_wpart[16];

  int tid = threadIdx.x;
  int bh = blockIdx.x;        // b*4 + h
  int h = bh & 3;
  if (tid < HEADDIM) s_xiq[tid] = xi_in[bh * HEADDIM + tid];
  __syncthreads();

  float bt = beta[h] * 0.5f;  // z = score / 2  (entmax15 pre-scale)
  float* z = zbuf + (size_t)bh * NDOCS;
  float lmax = -3.0e38f;
  for (int n = tid; n < NDOCS; n += 1024) {
    const v16bf* kr = (const v16bf*)(Kb + (size_t)n * DPROJ + h * HEADDIM);
    if (n + 1024 < NDOCS)
      __builtin_prefetch(Kb + (size_t)(n + 1024) * DPROJ + h * HEADDIM, 0, 1);
    float dot = 0.f;
#pragma unroll
    for (int j = 0; j < 4; j++) {
      v16bf kv = kr[j];
#pragma unroll
      for (int e = 0; e < 16; e++) dot += (float)kv[e] * s_xiq[j * 16 + e];
    }
    float zz = dot * bt;
    z[n] = zz;
    lmax = fmaxf(lmax, zz);
  }
  s_red[tid] = lmax; __syncthreads();
  for (int off = 512; off > 0; off >>= 1) {
    if (tid < off) s_red[tid] = fmaxf(s_red[tid], s_red[tid + off]);
    __syncthreads();
  }
  float zmax = s_red[0];
  __syncthreads();

  // Bisect tau in [zmax-1, zmax]: S(tau) = sum max(z-tau,0)^2 is monotone.
  float lo = zmax - 1.0f, hi = zmax;
  for (int it = 0; it < 30; it++) {
    float tau = 0.5f * (lo + hi);
    float ps = 0.f;
    for (int n = tid; n < NDOCS; n += 1024) {
      float d = z[n] - tau;
      ps += (d > 0.f) ? d * d : 0.f;
    }
    s_red[tid] = ps; __syncthreads();
    for (int off = 512; off > 0; off >>= 1) {
      if (tid < off) s_red[tid] += s_red[tid + off];
      __syncthreads();
    }
    float S = s_red[0];
    __syncthreads();
    if (S > 1.f) lo = tau; else hi = tau;
  }
  float tau = 0.5f * (lo + hi);

  // Deterministic weighted sum: 16 n-chunks x 64 dims, fixed reduction order.
  int chunk = tid >> 6;   // 0..15
  int d = tid & 63;
  int n0 = chunk * (NDOCS / 16), n1 = n0 + (NDOCS / 16);
  float acc = 0.f, wsum = 0.f;
  for (int n = n0; n < n1; n++) {
    float dd = z[n] - tau;
    if (dd > 0.f) {
      float w2 = dd * dd;
      acc += w2 * (float)Vb[(size_t)n * DPROJ + h * HEADDIM + d];
      wsum += w2;
    }
  }
  s_part[chunk][d] = acc;
  if (d == 0) s_wpart[chunk] = wsum;
  __syncthreads();
  if (tid < HEADDIM) {
    float a = 0.f, w = 0.f;
    for (int c2 = 0; c2 < 16; c2++) { a += s_part[c2][tid]; w += s_wpart[c2]; }
    xi_out[bh * HEADDIM + tid] = a / fmaxf(w, 1e-30f);
  }
}

// ---------------------------------------------------------------------------
// Kernel 4: conv @ Wo.T (8x2048) -> ws, and conv @ W_retrieval.T -> d_out tail
// ---------------------------------------------------------------------------
__global__ __launch_bounds__(256) void hop_out_proj(
    const float* __restrict__ conv, const float* __restrict__ Wo,
    const float* __restrict__ Wr, float* __restrict__ injp,
    float* __restrict__ retr)
{
  __shared__ float s_c[BATCH * DPROJ];
  int tid = threadIdx.x;
  for (int i = tid; i < BATCH * DPROJ; i += 256) s_c[i] = conv[i];
  __syncthreads();
  int gid = blockIdx.x * 256 + tid;
  if (gid < BATCH * HID) {
    int b = gid >> 11, j = gid & (HID - 1);
    const float* wr = Wo + (size_t)j * DPROJ;
    const float* cv = s_c + b * DPROJ;
    float acc = 0.f;
    for (int k = 0; k < DPROJ; k++) acc += cv[k] * wr[k];
    injp[gid] = acc;
  } else {
    int g2 = gid - BATCH * HID;
    if (g2 < BATCH * DMODEL) {
      int b = g2 / DMODEL, j = g2 - b * DMODEL;
      const float* wr = Wr + (size_t)j * DPROJ;
      const float* cv = s_c + b * DPROJ;
      float acc = 0.f;
      for (int k = 0; k < DPROJ; k++) acc += cv[k] * wr[k];
      retr[b * DMODEL + j] = acc;
    }
  }
}

// ---------------------------------------------------------------------------
// Kernel 5: broadcast (8x2048) over s=2048 -> 134 MB nontemporal store stream
// ---------------------------------------------------------------------------
__global__ __launch_bounds__(256) void hop_broadcast(
    const float* __restrict__ injp, float* __restrict__ out)
{
  size_t i4 = (size_t)blockIdx.x * 256 + threadIdx.x;
  size_t base = i4 * 4;
  size_t bs = base >> 11;          // b*SEQ + s
  int b = (int)(bs >> 11);
  int c = (int)(base & (size_t)(HID - 1));
  f4 v = *(const f4*)(injp + b * HID + c);
  __builtin_nontemporal_store(v, (f4*)(out + base));
}

// ---------------------------------------------------------------------------
extern "C" void kernel_launch(void* const* d_in, const int* in_sizes, int n_in,
                              void* d_out, int out_size, void* d_ws, size_t ws_size,
                              hipStream_t stream)
{
  const float* qe  = (const float*)d_in[1];
  const float* mem = (const float*)d_in[2];
  const float* Wq  = (const float*)d_in[3];
  const float* Wk  = (const float*)d_in[4];
  const float* Wv  = (const float*)d_in[5];
  const float* Wo  = (const float*)d_in[6];
  const float* Wr  = (const float*)d_in[7];
  const float* lb  = (const float*)d_in[8];
  const float* gq  = (const float*)d_in[9];
  const float* bq  = (const float*)d_in[10];
  const float* gm  = (const float*)d_in[11];
  const float* bm  = (const float*)d_in[12];
  (void)in_sizes; (void)n_in; (void)out_size; (void)ws_size;

  char* w = (char*)d_ws;
  size_t off = 0;
  auto take = [&](size_t bytes) -> char* {
    char* p = w + off;
    off = (off + bytes + 255) & ~(size_t)255;
    return p;
  };
  void*  Kb   = (void*)take((size_t)NDOCS * DPROJ * 2);           // bf16 K, 51.2 MB
  void*  Vb   = (void*)take((size_t)NDOCS * DPROJ * 2);           // bf16 V, 51.2 MB
  float* zbuf = (float*)take((size_t)BATCH * NHEADS * NDOCS * 4); // 12.8 MB
  float* xiA  = (float*)take((size_t)BATCH * DPROJ * 4);
  float* xiB  = (float*)take((size_t)BATCH * DPROJ * 4);
  float* beta = (float*)take(256);
  float* injp = (float*)take((size_t)BATCH * HID * 4);

  float* out_inj  = (float*)d_out;
  float* out_retr = out_inj + (size_t)BATCH * SEQ * HID;

  hipLaunchKernelGGL(hop_prep, dim3(1), dim3(256), 0, stream,
                     qe, Wq, gq, bq, lb, xiA, beta);
  hipLaunchKernelGGL(hop_kv_project, dim3(NDOCS / TILE_DOCS), dim3(256),
                     KV_DYN_LDS, stream, mem, Wk, Wv, gm, bm, Kb, Vb);
  hipLaunchKernelGGL(hop_step, dim3(BATCH * NHEADS), dim3(1024), 0, stream,
                     Kb, Vb, beta, xiA, xiB, zbuf);
  hipLaunchKernelGGL(hop_step, dim3(BATCH * NHEADS), dim3(1024), 0, stream,
                     Kb, Vb, beta, xiB, xiA, zbuf);
  hipLaunchKernelGGL(hop_step, dim3(BATCH * NHEADS), dim3(1024), 0, stream,
                     Kb, Vb, beta, xiA, xiB, zbuf);
  hipLaunchKernelGGL(hop_out_proj, dim3(88), dim3(256), 0, stream,
                     xiB, Wo, Wr, injp, out_retr);
  hipLaunchKernelGGL(hop_broadcast,
                     dim3((BATCH * SEQ * HID) / (256 * 4)), dim3(256), 0, stream,
                     injp, out_inj);
}
